// CausalSelfAttention_68805376082477
// MI455X (gfx1250) — compile-verified, matrix operand order checked by reading, not run on hardware
//
#include <hip/hip_runtime.h>
#include <hip/hip_bf16.h>

typedef __attribute__((ext_vector_type(16))) _Float16 v16h;
typedef __attribute__((ext_vector_type(8)))  _Float16 v8h;
typedef __attribute__((ext_vector_type(8)))  float    v8f;

#define NHEAD 12
#define TSEQ  2048
#define CDIM  768
#define HS    64
#define MTOT  4096   // B*T
#define N3C   2304

// Load a 16x32 (MxK) f16 A-fragment (or, by symmetry, a 32x16 B-fragment from a
// transposed [N][K] layout) from a row-major tile with leading dim `ldk`.
// Per ISA 7.12.2: lanes 0-15 hold row (lane), K in {0..7,16..23};
// lanes 16-31 hold row (lane-16), K in {8..15,24..31}.
__device__ __forceinline__ v16h load_frag16(const _Float16* base, int ldk) {
    const int lane = threadIdx.x & 31;
    const int row  = lane & 15;
    const int k0   = (lane & 16) ? 8 : 0;
    const _Float16* p = base + row * ldk + k0;
    v8h lo = *(const v8h*)(p);        // K = k0 .. k0+7   (16B aligned)
    v8h hi = *(const v8h*)(p + 16);   // K = k0+16 .. k0+23
    v16h r;
#pragma unroll
    for (int i = 0; i < 8; ++i) { r[i] = lo[i]; r[8 + i] = hi[i]; }
    return r;
}

__device__ __forceinline__ v8f wmma_f16(v16h a, v16h b, v8f c) {
    return __builtin_amdgcn_wmma_f32_16x16x32_f16(false, a, false, b, (short)0, c, false, false);
}

// Software-pipelined 32x64 GEMM strip: acc[mf][nf] += A[32xK] * B^T[64xK]^T.
// Fully unrolled, unroll-by-2 double buffering: cur/nxt fragment sets become
// per-step SSA values (no register-rotation movs); loads for the next k-step
// are issued before the current step's WMMAs so they overlap.
__device__ __forceinline__ void gemm_strip_32x64(const _Float16* abase,
                                                 const _Float16* bbase,
                                                 v8f acc[2][4]) {
    v16h ca0 = load_frag16(abase, CDIM);
    v16h ca1 = load_frag16(abase + 16 * CDIM, CDIM);
    v16h cb[4];
#pragma unroll
    for (int nf = 0; nf < 4; ++nf)
        cb[nf] = load_frag16(bbase + nf * 16 * CDIM, CDIM);

#pragma unroll
    for (int k = 0; k < CDIM; k += 64) {
        v16h na0, na1, nb[4];
        if (k + 32 < CDIM) {   // compile-time constant per unrolled step
            na0 = load_frag16(abase + k + 32, CDIM);
            na1 = load_frag16(abase + 16 * CDIM + k + 32, CDIM);
#pragma unroll
            for (int nf = 0; nf < 4; ++nf)
                nb[nf] = load_frag16(bbase + nf * 16 * CDIM + k + 32, CDIM);
        }
#pragma unroll
        for (int nf = 0; nf < 4; ++nf) {
            acc[0][nf] = wmma_f16(ca0, cb[nf], acc[0][nf]);
            acc[1][nf] = wmma_f16(ca1, cb[nf], acc[1][nf]);
        }
        if (k + 64 < CDIM) {
            ca0 = load_frag16(abase + k + 64, CDIM);
            ca1 = load_frag16(abase + 16 * CDIM + k + 64, CDIM);
#pragma unroll
            for (int nf = 0; nf < 4; ++nf)
                cb[nf] = load_frag16(bbase + nf * 16 * CDIM + k + 64, CDIM);
        }
#pragma unroll
        for (int nf = 0; nf < 4; ++nf) {
            acc[0][nf] = wmma_f16(na0, nb[nf], acc[0][nf]);
            acc[1][nf] = wmma_f16(na1, nb[nf], acc[1][nf]);
        }
    }
}

// ---------------- conversion / transpose pre-pass ----------------
__global__ void __launch_bounds__(256) cvt_f16_kernel(_Float16* dst, const float* src, int n) {
    int i = blockIdx.x * 256 + threadIdx.x;
    if (i < n) dst[i] = (_Float16)src[i];
}

// dst[N][K] = (f16) src[K][N]; 2D grid (no div/mod): y = source row k, x sweeps n
__global__ void __launch_bounds__(256) transpose_cvt_kernel(_Float16* dst, const float* src, int K, int N) {
    const int n = blockIdx.x * 256 + threadIdx.x;
    const int k = blockIdx.y;
    if (n < N) dst[n * K + k] = (_Float16)src[k * N + n];
}

// ---------------- QKV GEMM: [4096x768] x [768x2304] + bias -> Q,K,Vt (f16) ----
__global__ void __launch_bounds__(256) qkv_gemm_kernel(
    const _Float16* __restrict__ xh,   // [4096][768]
    const _Float16* __restrict__ wt,   // [2304][768] (transposed W_attn)
    const float*    __restrict__ bias, // [2304]
    _Float16* __restrict__ Qh,         // [24][2048][64], pre-scaled by 1/8
    _Float16* __restrict__ Kh,         // [24][2048][64]
    _Float16* __restrict__ Vt)         // [24][64][2048]
{
    const int lane = threadIdx.x & 31;
    const int wave = threadIdx.x >> 5;
    const int m0 = blockIdx.x * 256 + wave * 32;   // 32 rows per wave
    const int n0 = blockIdx.y * 64;

    v8f acc[2][4] = {};
    gemm_strip_32x64(xh + m0 * CDIM, wt + n0 * CDIM, acc);

    const int nlo = lane & 15;
    const int mhi = (lane & 16) ? 8 : 0;
#pragma unroll
    for (int mf = 0; mf < 2; ++mf) {
#pragma unroll
        for (int nf = 0; nf < 4; ++nf) {
            const int n = n0 + nf * 16 + nlo;
            const float bv = bias[n];
#pragma unroll
            for (int r = 0; r < 8; ++r) {
                const int m = m0 + mf * 16 + r + mhi;
                const float v = acc[mf][nf][r] + bv;
                const int b = m >> 11, t = m & (TSEQ - 1);
                if (n < CDIM) {                     // Q (apply 1/sqrt(64))
                    const int h = n >> 6, d = n & 63;
                    Qh[((b * NHEAD + h) * TSEQ + t) * HS + d] = (_Float16)(v * 0.125f);
                } else if (n < 2 * CDIM) {          // K
                    const int nn = n - CDIM, h = nn >> 6, d = nn & 63;
                    Kh[((b * NHEAD + h) * TSEQ + t) * HS + d] = (_Float16)v;
                } else {                            // V, stored transposed [d][t]
                    const int nn = n - 2 * CDIM, h = nn >> 6, d = nn & 63;
                    Vt[((b * NHEAD + h) * HS + d) * TSEQ + t] = (_Float16)v;
                }
            }
        }
    }
}

// ---------------- causal flash attention ----------------
__global__ void __launch_bounds__(128) attn_kernel(
    const _Float16* __restrict__ Qh,
    const _Float16* __restrict__ Kh,
    const _Float16* __restrict__ Vt,
    _Float16* __restrict__ yh)         // [4096][768]
{
    __shared__ _Float16 pbuf[4 * 16 * 64];   // per-wave private 16x64 P tile
    const int lane = threadIdx.x & 31;
    const int wave = threadIdx.x >> 5;
    const int qt = blockIdx.x;          // query tile (64 rows)
    const int bh = blockIdx.y;          // b*NHEAD + h
    const int q0 = qt * 64;
    const int qm = q0 + wave * 16;      // this wave's first query row
    const int nlo = lane & 15;
    const int mhi = (lane & 16) ? 8 : 0;

    const _Float16* Qbase = Qh + (bh * TSEQ + qm) * HS;
    const v16h aq0 = load_frag16(Qbase + 0, HS);
    const v16h aq1 = load_frag16(Qbase + 32, HS);

    v8f o[4] = {};
    float mi[8], li[8];
#pragma unroll
    for (int r = 0; r < 8; ++r) { mi[r] = -1e30f; li[r] = 0.0f; }

    _Float16* pw = pbuf + wave * 16 * 64;   // only this wave touches this region

    for (int kt = 0; kt <= q0; kt += 64) {
        // ---- load ALL K fragments for this key tile, then 8 back-to-back WMMAs
        v16h kb[4][2];
#pragma unroll
        for (int nf = 0; nf < 4; ++nf) {
            const _Float16* Kb = Kh + (bh * TSEQ + kt + nf * 16) * HS;
            kb[nf][0] = load_frag16(Kb, HS);
            kb[nf][1] = load_frag16(Kb + 32, HS);
        }
        v8f s[4];
#pragma unroll
        for (int nf = 0; nf < 4; ++nf) {
            v8f z = {};
            z = wmma_f16(aq0, kb[nf][0], z);
            s[nf] = wmma_f16(aq1, kb[nf][1], z);
        }

        // ---- issue V-fragment loads now; softmax VALU work hides their latency
        v16h vb[4][2];
#pragma unroll
        for (int nf = 0; nf < 4; ++nf) {
            const _Float16* Vb = Vt + (bh * HS + nf * 16) * TSEQ + kt;
            vb[nf][0] = load_frag16(Vb, TSEQ);
            vb[nf][1] = load_frag16(Vb + 32, TSEQ);
        }

        // ---- causal mask (only the diagonal tile can violate j <= q)
        if (kt == q0) {
#pragma unroll
            for (int nf = 0; nf < 4; ++nf)
#pragma unroll
                for (int r = 0; r < 8; ++r) {
                    const int j = kt + nf * 16 + nlo;
                    const int q = qm + r + mhi;
                    if (j > q) s[nf][r] = -1e30f;
                }
        }

        // ---- online softmax; row r lives in one 16-lane half-group per VGPR
        float alpha[8];
#pragma unroll
        for (int r = 0; r < 8; ++r) {
            float x = fmaxf(fmaxf(s[0][r], s[1][r]), fmaxf(s[2][r], s[3][r]));
            x = fmaxf(x, __shfl_xor(x, 1, 32));
            x = fmaxf(x, __shfl_xor(x, 2, 32));
            x = fmaxf(x, __shfl_xor(x, 4, 32));
            x = fmaxf(x, __shfl_xor(x, 8, 32));
            const float mnew = fmaxf(mi[r], x);
            alpha[r] = __expf(mi[r] - mnew);
            mi[r] = mnew;
            float rs = 0.0f;
#pragma unroll
            for (int nf = 0; nf < 4; ++nf) {
                const float p = __expf(s[nf][r] - mnew);
                s[nf][r] = p;
                rs += p;
            }
            rs += __shfl_xor(rs, 1, 32);
            rs += __shfl_xor(rs, 2, 32);
            rs += __shfl_xor(rs, 4, 32);
            rs += __shfl_xor(rs, 8, 32);
            li[r] = li[r] * alpha[r] + rs;
        }
#pragma unroll
        for (int nf = 0; nf < 4; ++nf)
#pragma unroll
            for (int r = 0; r < 8; ++r)
                o[nf][r] *= alpha[r];

        // ---- P (C-layout f32) -> per-wave LDS tile -> reload as A-fragments.
        // No block barrier needed: each wave reads only its own region; same-wave
        // LDS store->load ordering is enforced via DScnt waits.
#pragma unroll
        for (int nf = 0; nf < 4; ++nf)
#pragma unroll
            for (int r = 0; r < 8; ++r)
                pw[(r + mhi) * 64 + nf * 16 + nlo] = (_Float16)s[nf][r];

        const v16h pa0 = load_frag16(pw, 64);
        const v16h pa1 = load_frag16(pw + 32, 64);
#pragma unroll
        for (int nf = 0; nf < 4; ++nf) {
            o[nf] = wmma_f16(pa0, vb[nf][0], o[nf]);
            o[nf] = wmma_f16(pa1, vb[nf][1], o[nf]);
        }
    }

    // ---- normalize + write y (f16, row-major [b*T+t][h*64+d])
    const int b = bh / NHEAD, h = bh % NHEAD;
#pragma unroll
    for (int r = 0; r < 8; ++r) {
        const float inv = 1.0f / li[r];
        const int t = qm + r + mhi;
#pragma unroll
        for (int nf = 0; nf < 4; ++nf) {
            const int col = h * HS + nf * 16 + nlo;
            yh[(b * TSEQ + t) * CDIM + col] = (_Float16)(o[nf][r] * inv);
        }
    }
}

// ---------------- output projection GEMM ----------------
__global__ void __launch_bounds__(256) proj_gemm_kernel(
    const _Float16* __restrict__ yh,   // [4096][768]
    const _Float16* __restrict__ wt,   // [768][768] (transposed W_proj)
    const float*    __restrict__ bias, // [768]
    float* __restrict__ out)           // [4096][768]
{
    const int lane = threadIdx.x & 31;
    const int wave = threadIdx.x >> 5;
    const int m0 = blockIdx.x * 256 + wave * 32;
    const int n0 = blockIdx.y * 64;

    v8f acc[2][4] = {};
    gemm_strip_32x64(yh + m0 * CDIM, wt + n0 * CDIM, acc);

    const int nlo = lane & 15;
    const int mhi = (lane & 16) ? 8 : 0;
#pragma unroll
    for (int mf = 0; mf < 2; ++mf)
#pragma unroll
        for (int nf = 0; nf < 4; ++nf) {
            const int n = n0 + nf * 16 + nlo;
            const float bv = bias[n];
#pragma unroll
            for (int r = 0; r < 8; ++r)
                out[(m0 + mf * 16 + r + mhi) * CDIM + n] = acc[mf][nf][r] + bv;
        }
}

extern "C" void kernel_launch(void* const* d_in, const int* in_sizes, int n_in,
                              void* d_out, int out_size, void* d_ws, size_t ws_size,
                              hipStream_t stream) {
    const float* x      = (const float*)d_in[0];
    const float* W_attn = (const float*)d_in[1];
    const float* b_attn = (const float*)d_in[2];
    const float* W_proj = (const float*)d_in[3];
    const float* b_proj = (const float*)d_in[4];
    float* out = (float*)d_out;

    char* w = (char*)d_ws;
    size_t off = 0;
    auto carve = [&](size_t bytes) {
        char* p = w + off;
        off += (bytes + 255) & ~(size_t)255;
        return p;
    };
    _Float16* xh  = (_Float16*)carve((size_t)MTOT * CDIM * 2);        // 6.3 MB
    _Float16* wta = (_Float16*)carve((size_t)N3C * CDIM * 2);         // 3.5 MB
    _Float16* wtp = (_Float16*)carve((size_t)CDIM * CDIM * 2);        // 1.2 MB
    _Float16* Qh  = (_Float16*)carve((size_t)2 * NHEAD * TSEQ * HS * 2);
    _Float16* Kh  = (_Float16*)carve((size_t)2 * NHEAD * TSEQ * HS * 2);
    _Float16* Vt  = (_Float16*)carve((size_t)2 * NHEAD * TSEQ * HS * 2);
    _Float16* yh  = (_Float16*)carve((size_t)MTOT * CDIM * 2);        // total ~36 MB

    const int nx = MTOT * CDIM;
    cvt_f16_kernel<<<(nx + 255) / 256, 256, 0, stream>>>(xh, x, nx);
    transpose_cvt_kernel<<<dim3((N3C + 255) / 256, CDIM), 256, 0, stream>>>(wta, W_attn, CDIM, N3C);
    transpose_cvt_kernel<<<dim3((CDIM + 255) / 256, CDIM), 256, 0, stream>>>(wtp, W_proj, CDIM, CDIM);

    qkv_gemm_kernel<<<dim3(MTOT / 256, N3C / 64), 256, 0, stream>>>(xh, wta, b_attn, Qh, Kh, Vt);
    attn_kernel<<<dim3(TSEQ / 64, 2 * NHEAD), 128, 0, stream>>>(Qh, Kh, Vt, yh);
    proj_gemm_kernel<<<dim3(MTOT / 256, CDIM / 64), 256, 0, stream>>>(yh, wtp, b_proj, out);
}
